// SoftRank_37194416783875
// MI455X (gfx1250) — compile-verified, matrix operand order checked by reading, not run on hardware
//
#include <hip/hip_runtime.h>

#define NN 2048
#define DD 8
#define BB 4
#define STRIDE 2052  // 2048 + 4 floats pad: kills LDS bank conflicts on the transpose fill

typedef __attribute__((ext_vector_type(16))) _Float16 v16h;
typedef __attribute__((ext_vector_type(8)))  float    v8f;

// sigmoid(1000*z) = 0.5 + 0.5*tanh(500*z)  -> one TRANS op per element on gfx1250
#if __has_builtin(__builtin_amdgcn_tanhf)
__device__ __forceinline__ float fast_tanh(float x) { return __builtin_amdgcn_tanhf(x); }
#elif __has_builtin(__builtin_amdgcn_tanh_f32)
__device__ __forceinline__ float fast_tanh(float x) { return __builtin_amdgcn_tanh_f32(x); }
#else
__device__ __forceinline__ float fast_tanh(float x) {
  // overflow-safe: exp(+inf) -> 1 - 0 = 1 ; exp(0) -> 1 - 2 = -1
  float e = __expf(2.0f * x);
  return 1.0f - 2.0f / (e + 1.0f);
}
#endif

__global__ __launch_bounds__(256)
void softrank_kernel(const float* __restrict__ x, float* __restrict__ out) {
  __shared__ __align__(16) float lds[DD * STRIDE];

  const int b    = blockIdx.x >> 7;          // 128 row-tiles per batch
  const int row0 = (blockIdx.x & 127) << 4;  // 16 output rows per block
  const float* xb = x + b * NN * DD;

  // Stage x[b,:,:] into LDS, transposed to column-major-per-d.
  // Padded stride (2052 % 64 == 4) -> 32 lanes hit 32 distinct banks.
  for (int e = (int)threadIdx.x; e < NN * DD; e += 256) {
    lds[(e & 7) * STRIDE + (e >> 3)] = xb[e];
  }
  __syncthreads();

  const int d       = threadIdx.x >> 5;        // one wave (wave32) per feature column
  const int lane    = threadIdx.x & 31;
  const int laneOff = (lane >> 4) << 3;        // A-matrix K split: low lanes K 0-7/16-23, high 8-15/24-31
  const float* col  = &lds[d * STRIDE];
  const float  xn   = col[row0 + (lane & 15)]; // this lane's row value (M = lane&15)

  v16h bones;
  #pragma unroll
  for (int j = 0; j < 16; ++j) bones[j] = (_Float16)1.0f;  // B = ones -> WMMA does the Sum_i reduction

  v8f acc = {};  // f32 C/D accumulator: Sum_i tanh(500*(x_n - x_i))

  #pragma unroll 2
  for (int it = 0; it < NN / 32; ++it) {
    const int i0 = it * 32 + laneOff;
    // 16 comparison values in ISA A-layout order via 4 aligned b128 LDS broadcasts
    float q[16];
    *(float4*)&q[0]  = *(const float4*)&col[i0];
    *(float4*)&q[4]  = *(const float4*)&col[i0 + 4];
    *(float4*)&q[8]  = *(const float4*)&col[i0 + 16];
    *(float4*)&q[12] = *(const float4*)&col[i0 + 20];

    v16h a;
    #pragma unroll
    for (int j = 0; j < 16; ++j) {
      a[j] = (_Float16)fast_tanh((xn - q[j]) * 500.0f);  // |tanh|<=1: f16-safe
    }
    // D = A(16x32 f16) x ones(32x16) + C  : row-sum reduction on the matrix pipe
    acc = __builtin_amdgcn_wmma_f32_16x16x32_f16(false, a, false, bones,
                                                 (short)0, acc, false, false);
  }

  // C/D layout: VGPR r = row r (lanes 0-15) / row 8+r (lanes 16-31); all 16 columns equal.
  if ((lane & 15) == 0) {
    const int mbase = (lane >> 4) << 3;
    #pragma unroll
    for (int r = 0; r < 8; ++r) {
      const int n = row0 + mbase + r;
      out[(b * NN + n) * DD + d] = 0.5f + acc[r] * (0.5f / (float)NN);
    }
  }
}

extern "C" void kernel_launch(void* const* d_in, const int* in_sizes, int n_in,
                              void* d_out, int out_size, void* d_ws, size_t ws_size,
                              hipStream_t stream) {
  const float* x  = (const float*)d_in[0];
  float* out      = (float*)d_out;
  (void)in_sizes; (void)n_in; (void)out_size; (void)d_ws; (void)ws_size;
  softrank_kernel<<<dim3(BB * (NN / 16)), dim3(256), 0, stream>>>(x, out);
}
